// Qwen3NextLinearAttention_60275571032319
// MI455X (gfx1250) — matrix operand order repaired and checked by reading, not compiled, not run on hardware
//
#include <hip/hip_runtime.h>

typedef __bf16 bf16;
typedef __attribute__((ext_vector_type(16))) __bf16 bf16x16;
typedef __attribute__((ext_vector_type(8)))  float  floatx8;

#define H_SIZE  2048
#define S_LEN   1024
#define BATCH   2
#define NTOK    (BATCH * S_LEN)     // 2048 tokens
#define KDIM    2048
#define VDIM    4096
#define QKVZ_N  12288
#define CONV_DIM 8192

// ---------------------------------------------------------------- converts
__global__ void cvt_bf16(const float* __restrict__ in, bf16* __restrict__ out, int n) {
    int i = blockIdx.x * blockDim.x + threadIdx.x;
    if (i < n) out[i] = (bf16)in[i];
}

// [R][C] f32 -> [C][R] bf16
__global__ void cvt_tr_bf16(const float* __restrict__ in, bf16* __restrict__ out, int R, int C) {
    int i = blockIdx.x * blockDim.x + threadIdx.x;
    if (i < R * C) {
        int r = i / C, c = i % C;
        out[(size_t)c * R + r] = (bf16)in[i];
    }
}

// ---------------------------------------------------------------- WMMA GEMM
// C[M,N] f32 = A[M,K]bf16 * B[N,K]bf16^T.  Block tile 128x128, BK=32.
// Double-buffered LDS; tiles staged with global_load_async_to_lds_b128 (ASYNCcnt),
// one barrier per K-step. 8 waves: wave grid 4(M) x 2(N); per-wave 2x4 wmma tiles.
__global__ __launch_bounds__(256) void gemm_bf16_wmma(
    const bf16* __restrict__ A, const bf16* __restrict__ B, float* __restrict__ C,
    int M, int N, int K) {
    __shared__ bf16 As[2][128][32];
    __shared__ bf16 Bs[2][128][32];
    const int tid   = threadIdx.x;
    const int lane  = tid & 31;
    const int wave  = tid >> 5;
    const int wmOff = (wave & 3) * 32;
    const int wnOff = (wave >> 2) * 64;
    const int bm    = blockIdx.y * 128;
    const int bn    = blockIdx.x * 128;

    floatx8 zero = {};
    floatx8 acc[2][4];
    for (int i = 0; i < 2; i++)
        for (int j = 0; j < 4; j++) acc[i][j] = zero;

    const int lr = tid >> 2;        // 0..63 row within tile
    const int lc = (tid & 3) * 8;   // 0,8,16,24 col chunk (8 bf16 = 16B)

    const bf16* aSrc0 = A + (size_t)(bm + lr) * K + lc;
    const bf16* aSrc1 = A + (size_t)(bm + lr + 64) * K + lc;
    const bf16* bSrc0 = B + (size_t)(bn + lr) * K + lc;
    const bf16* bSrc1 = B + (size_t)(bn + lr + 64) * K + lc;

    // async copy of one 128x32 A tile + 128x32 B tile into LDS buffer p
    auto issue_tile = [&](int p, int k0) {
        unsigned dA0 = (unsigned)(size_t)&As[p][lr][lc];
        unsigned dA1 = (unsigned)(size_t)&As[p][lr + 64][lc];
        unsigned dB0 = (unsigned)(size_t)&Bs[p][lr][lc];
        unsigned dB1 = (unsigned)(size_t)&Bs[p][lr + 64][lc];
        const bf16* s0 = aSrc0 + k0;
        const bf16* s1 = aSrc1 + k0;
        const bf16* s2 = bSrc0 + k0;
        const bf16* s3 = bSrc1 + k0;
        asm volatile("global_load_async_to_lds_b128 %0, %1, off" :: "v"(dA0), "v"(s0) : "memory");
        asm volatile("global_load_async_to_lds_b128 %0, %1, off" :: "v"(dA1), "v"(s1) : "memory");
        asm volatile("global_load_async_to_lds_b128 %0, %1, off" :: "v"(dB0), "v"(s2) : "memory");
        asm volatile("global_load_async_to_lds_b128 %0, %1, off" :: "v"(dB1), "v"(s3) : "memory");
    };

    issue_tile(0, 0);
    asm volatile("s_wait_asynccnt 0x0" ::: "memory");
    __syncthreads();

    const int hsel = lane >> 4;   // lane half
    const int rsel = lane & 15;

    for (int k0 = 0; k0 < K; k0 += 32) {
        const int cur = (k0 >> 5) & 1;
        if (k0 + 32 < K) issue_tile(cur ^ 1, k0 + 32);

        union Frag { bf16x16 v; uint4 u[2]; };
        Frag afrag[2], bfrag[4];
        // A 16-bit layout: lanes 0-15 hold K 0..7 / 16..23, lanes 16-31 K 8..15 / 24..31
        #pragma unroll
        for (int i = 0; i < 2; i++) {
            int row = wmOff + i * 16 + rsel;
            afrag[i].u[0] = *(const uint4*)&As[cur][row][hsel * 8];
            afrag[i].u[1] = *(const uint4*)&As[cur][row][16 + hsel * 8];
        }
        // B 16-bit layout: lanes 0-15 hold K 0..7 / 16..23, lanes 16-31 K 8..15 / 24..31
        #pragma unroll
        for (int j = 0; j < 4; j++) {
            int row = wnOff + j * 16 + rsel;
            bfrag[j].u[0] = *(const uint4*)&Bs[cur][row][hsel * 8];
            bfrag[j].u[1] = *(const uint4*)&Bs[cur][row][16 + hsel * 8];
        }
        #pragma unroll
        for (int i = 0; i < 2; i++)
            #pragma unroll
            for (int j = 0; j < 4; j++)
                acc[i][j] = __builtin_amdgcn_wmma_f32_16x16x32_bf16(
                    false, afrag[i].v, false, bfrag[j].v, (short)0, acc[i][j], false, false);

        asm volatile("s_wait_asynccnt 0x0" ::: "memory");
        __syncthreads();
    }

    const int cn = lane & 15;
    #pragma unroll
    for (int i = 0; i < 2; i++)
        #pragma unroll
        for (int j = 0; j < 4; j++) {
            int row0 = bm + wmOff + i * 16 + hsel * 8;
            int col  = bn + wnOff + j * 16 + cn;
            float* cp = C + (size_t)row0 * N + col;
            #pragma unroll
            for (int r = 0; r < 8; r++) cp[(size_t)r * N] = acc[i][j][r];
        }
}

// ---------------------------------------------------------------- small ba GEMM (N=64)
__global__ void ba_gemm(const float* __restrict__ H, const float* __restrict__ Wba,
                        float* __restrict__ BA) {
    int t = blockIdx.x;      // token
    int n = threadIdx.x;     // 64 columns
    const float* hrow = H + (size_t)t * H_SIZE;
    float acc = 0.f;
    for (int k = 0; k < H_SIZE; ++k) acc += hrow[k] * Wba[k * 64 + n];
    BA[t * 64 + n] = acc;
}

// ---------------------------------------------------------------- conv channel remap
__device__ __forceinline__ int qkvz_channel(int cc) {
    if (cc < 2048) { int h = cc >> 7;             return h * 768 + (cc & 127); }
    if (cc < 4096) { int c = cc - 2048; int h = c >> 7; return h * 768 + 128 + (c & 127); }
    int c = cc - 4096; int hv = c >> 7; int dv = c & 127;
    return (hv >> 1) * 768 + 256 + ((hv & 1) << 7) + dv;
}

// depthwise causal conv K=4 + silu, scatter into q/k/v buffers
__global__ void conv_silu(const float* __restrict__ qkvz, const float* __restrict__ cw,
                          float* __restrict__ q, float* __restrict__ k, float* __restrict__ v) {
    int idx = blockIdx.x * blockDim.x + threadIdx.x;   // NTOK * 8192
    int cc  = idx & 8191;
    int bs  = idx >> 13;
    int s   = bs & (S_LEN - 1);
    const float* base = qkvz + (size_t)bs * QKVZ_N + qkvz_channel(cc);
    float w0 = cw[cc * 4 + 0], w1 = cw[cc * 4 + 1], w2 = cw[cc * 4 + 2], w3 = cw[cc * 4 + 3];
    float acc = base[0] * w3;
    if (s >= 1) acc += base[-(int)QKVZ_N]     * w2;
    if (s >= 2) acc += base[-2 * (int)QKVZ_N] * w1;
    if (s >= 3) acc += base[-3 * (int)QKVZ_N] * w0;
    acc = acc / (1.f + expf(-acc));   // silu
    if (cc < 2048)       q[(size_t)bs * KDIM + cc]          = acc;
    else if (cc < 4096)  k[(size_t)bs * KDIM + (cc - 2048)] = acc;
    else                 v[(size_t)bs * VDIM + (cc - 4096)] = acc;
}

// ---------------------------------------------------------------- l2norm q/k (per head of 128)
__global__ void l2norm_qk(float* __restrict__ q, float* __restrict__ k) {
    int bs = blockIdx.x, y = blockIdx.y, t = threadIdx.x;    // 128 threads
    float* buf = (y < 16) ? (q + ((size_t)bs * 16 + y) * 128)
                          : (k + ((size_t)bs * 16 + (y - 16)) * 128);
    float val = buf[t];
    float ss  = val * val;
    #pragma unroll
    for (int d = 16; d; d >>= 1) ss += __shfl_xor(ss, d, 32);
    __shared__ float red[4];
    if ((t & 31) == 0) red[t >> 5] = ss;
    __syncthreads();
    float tot = red[0] + red[1] + red[2] + red[3];
    float scale = rsqrtf(tot + 1e-6f);
    if (y < 16) scale *= 0.08838834764831845f;   // DK^-0.5
    buf[t] = val * scale;
}

// ---------------------------------------------------------------- beta / g
__global__ void beta_g(const float* __restrict__ BA, const float* __restrict__ dt_bias,
                       const float* __restrict__ A_log, float* __restrict__ beta,
                       float* __restrict__ g) {
    int i  = blockIdx.x * blockDim.x + threadIdx.x;   // NTOK*32
    int vh = i & 31;
    int bs = i >> 5;
    int h = vh >> 1, r = vh & 1;
    float bval = BA[bs * 64 + h * 4 + r];
    float aval = BA[bs * 64 + h * 4 + 2 + r];
    beta[i] = 1.f / (1.f + expf(-bval));
    float x  = aval + dt_bias[vh];
    float sp = (x > 20.f) ? x : log1pf(expf(x));
    g[i] = -expf(A_log[vh]) * sp;
}

// ---------------------------------------------------------------- gated delta recurrence
// 256 blocks: (b, head, 32-wide v-chunk). 8 waves x 4 v-cols; lane owns 4 k-rows.
__global__ __launch_bounds__(256) void gdn_recurrence(
    const float* __restrict__ q, const float* __restrict__ k, const float* __restrict__ v,
    const float* __restrict__ beta, const float* __restrict__ g, float* __restrict__ out) {
    int bi   = blockIdx.x;
    int vblk = bi & 3;
    int h    = (bi >> 2) & 31;
    int b    = bi >> 7;
    int lane = threadIdx.x & 31;
    int w    = threadIdx.x >> 5;
    int kh   = h >> 1;                       // GQA: 32 v-heads share 16 k-heads
    int vbase = vblk * 32 + w * 4;

    float st[4][4];
    #pragma unroll
    for (int j = 0; j < 4; j++)
        #pragma unroll
        for (int i = 0; i < 4; i++) st[j][i] = 0.f;

    const float* qp = q + ((size_t)b * S_LEN * 16 + kh) * 128;
    const float* kp = k + ((size_t)b * S_LEN * 16 + kh) * 128;
    const float* vp = v + ((size_t)b * S_LEN * 32 + h) * 128 + vbase;
    const float* gp = g    + (size_t)b * S_LEN * 32 + h;
    const float* bp = beta + (size_t)b * S_LEN * 32 + h;
    float*       op = out + ((size_t)b * S_LEN * 32 + h) * 128 + vbase;

    for (int s = 0; s < S_LEN; ++s) {
        // keep the streaming loads ahead of the scan (global_prefetch_b8)
        if ((s & 7) == 0 && s + 8 < S_LEN) {
            __builtin_prefetch(kp + (size_t)(s + 8) * KDIM + lane, 0, 0);
            __builtin_prefetch(qp + (size_t)(s + 8) * KDIM + lane, 0, 0);
            __builtin_prefetch(vp + (size_t)(s + 8) * VDIM, 0, 0);
        }
        float dec = expf(gp[(size_t)s * 32]);
        float bt  = bp[(size_t)s * 32];
        float kreg[4], qreg[4], vreg[4];
        #pragma unroll
        for (int j = 0; j < 4; j++) {
            kreg[j] = kp[(size_t)s * KDIM + lane + 32 * j];
            qreg[j] = qp[(size_t)s * KDIM + lane + 32 * j];
        }
        #pragma unroll
        for (int i = 0; i < 4; i++) vreg[i] = vp[(size_t)s * VDIM + i];

        float kv[4] = {0.f, 0.f, 0.f, 0.f};
        #pragma unroll
        for (int j = 0; j < 4; j++)
            #pragma unroll
            for (int i = 0; i < 4; i++) { st[j][i] *= dec; kv[i] += st[j][i] * kreg[j]; }
        #pragma unroll
        for (int d = 16; d; d >>= 1)
            #pragma unroll
            for (int i = 0; i < 4; i++) kv[i] += __shfl_xor(kv[i], d, 32);

        float delta[4], o[4] = {0.f, 0.f, 0.f, 0.f};
        #pragma unroll
        for (int i = 0; i < 4; i++) delta[i] = (vreg[i] - kv[i]) * bt;
        #pragma unroll
        for (int j = 0; j < 4; j++)
            #pragma unroll
            for (int i = 0; i < 4; i++) { st[j][i] += kreg[j] * delta[i]; o[i] += st[j][i] * qreg[j]; }
        #pragma unroll
        for (int d = 16; d; d >>= 1)
            #pragma unroll
            for (int i = 0; i < 4; i++) o[i] += __shfl_xor(o[i], d, 32);

        if (lane == 0) {
            #pragma unroll
            for (int i = 0; i < 4; i++) op[(size_t)s * VDIM + i] = o[i];
        }
    }
}

// ---------------------------------------------------------------- gated RMSNorm -> bf16
__global__ void gated_rmsnorm(const float* __restrict__ attn, const float* __restrict__ qkvz,
                              const float* __restrict__ norm_w, bf16* __restrict__ Y) {
    int idx = blockIdx.x;            // NTOK * 32
    int vh  = idx & 31;
    int bs  = idx >> 5;
    int t   = threadIdx.x;           // 128
    float o = attn[(size_t)idx * 128 + t];
    int zoff = (vh >> 1) * 768 + 512 + ((vh & 1) << 7) + t;
    float z  = qkvz[(size_t)bs * QKVZ_N + zoff];
    float xg = o * (z / (1.f + expf(-z)));
    float ss = xg * xg;
    #pragma unroll
    for (int d = 16; d; d >>= 1) ss += __shfl_xor(ss, d, 32);
    __shared__ float red[4];
    if ((t & 31) == 0) red[t >> 5] = ss;
    __syncthreads();
    float tot = red[0] + red[1] + red[2] + red[3];
    float y = xg * rsqrtf(tot * (1.f / 128.f) + 1e-6f) * norm_w[t];
    Y[(size_t)bs * VDIM + vh * 128 + t] = (bf16)y;
}

// ---------------------------------------------------------------- launch
extern "C" void kernel_launch(void* const* d_in, const int* in_sizes, int n_in,
                              void* d_out, int out_size, void* d_ws, size_t ws_size,
                              hipStream_t stream) {
    const float* hidden  = (const float*)d_in[0];
    const float* w_qkvz  = (const float*)d_in[1];
    const float* w_ba    = (const float*)d_in[2];
    const float* conv_w  = (const float*)d_in[3];
    const float* dt_bias = (const float*)d_in[4];
    const float* A_log   = (const float*)d_in[5];
    const float* norm_w  = (const float*)d_in[6];
    const float* w_out   = (const float*)d_in[7];
    (void)in_sizes; (void)n_in; (void)out_size; (void)ws_size;

    char*  ws  = (char*)d_ws;
    size_t off = 0;
    auto take = [&](size_t bytes) {
        char* p = ws + off;
        off += (bytes + 255) & ~(size_t)255;
        return p;
    };
    bf16*  Abf  = (bf16*) take((size_t)NTOK  * H_SIZE * 2);
    bf16*  B1bf = (bf16*) take((size_t)QKVZ_N * H_SIZE * 2);
    bf16*  W2bf = (bf16*) take((size_t)H_SIZE * VDIM  * 2);
    float* qkvz = (float*)take((size_t)NTOK * QKVZ_N * 4);
    float* ba   = (float*)take((size_t)NTOK * 64 * 4);
    float* qb   = (float*)take((size_t)NTOK * KDIM * 4);
    float* kb   = (float*)take((size_t)NTOK * KDIM * 4);
    float* vb   = (float*)take((size_t)NTOK * VDIM * 4);
    float* beta = (float*)take((size_t)NTOK * 32 * 4);
    float* gbuf = (float*)take((size_t)NTOK * 32 * 4);
    float* attn = (float*)take((size_t)NTOK * VDIM * 4);
    bf16*  Ybf  = (bf16*) take((size_t)NTOK * VDIM * 2);

    cvt_bf16   <<<(NTOK * H_SIZE) / 256, 256, 0, stream>>>(hidden, Abf, NTOK * H_SIZE);
    cvt_tr_bf16<<<(H_SIZE * QKVZ_N) / 256, 256, 0, stream>>>(w_qkvz, B1bf, H_SIZE, QKVZ_N);
    cvt_tr_bf16<<<(VDIM * H_SIZE) / 256, 256, 0, stream>>>(w_out, W2bf, VDIM, H_SIZE);

    gemm_bf16_wmma<<<dim3(QKVZ_N / 128, NTOK / 128), 256, 0, stream>>>(
        Abf, B1bf, qkvz, NTOK, QKVZ_N, H_SIZE);
    ba_gemm<<<NTOK, 64, 0, stream>>>(hidden, w_ba, ba);

    conv_silu<<<(NTOK * CONV_DIM) / 256, 256, 0, stream>>>(qkvz, conv_w, qb, kb, vb);
    l2norm_qk<<<dim3(NTOK, 32), 128, 0, stream>>>(qb, kb);
    beta_g<<<(NTOK * 32) / 256, 256, 0, stream>>>(ba, dt_bias, A_log, beta, gbuf);

    gdn_recurrence<<<BATCH * 32 * 4, 256, 0, stream>>>(qb, kb, vb, beta, gbuf, attn);

    gated_rmsnorm<<<NTOK * 32, 128, 0, stream>>>(attn, qkvz, norm_w, Ybf);
    gemm_bf16_wmma<<<dim3(H_SIZE / 128, NTOK / 128), 256, 0, stream>>>(
        Ybf, W2bf, (float*)d_out, NTOK, H_SIZE, VDIM);
}
